// CausalMultiHeadSelfAttention_6691559047728
// MI455X (gfx1250) — matrix operand order repaired and checked by reading, not compiled, not run on hardware
//
#include <hip/hip_runtime.h>

#define S_LEN 2048
#define BATCH 4
#define DM    1024
#define NH    16
#define DKH   64

typedef _Float16 v16h __attribute__((ext_vector_type(16)));
typedef _Float16 v8h  __attribute__((ext_vector_type(8)));
typedef float    v8f  __attribute__((ext_vector_type(8)));
typedef unsigned int u32x4 __attribute__((ext_vector_type(4)));
typedef int          i32x4 __attribute__((ext_vector_type(4)));
typedef int          i32x8 __attribute__((ext_vector_type(8)));

__device__ __forceinline__ v16h frag16(const _Float16* p0, const _Float16* p1) {
    union { v16h v; v8h h[2]; } u;
    u.h[0] = *(const v8h*)p0;
    u.h[1] = *(const v8h*)p1;
    return u.v;
}

__device__ __forceinline__ v8f wmma_f16(v16h a, v16h b, v8f c) {
    return __builtin_amdgcn_wmma_f32_16x16x32_f16(false, a, false, b, (short)0, c, false, false);
}

// ---------------------------------------------------------------------------
// Tensor Data Mover: async DMA of a 2D f16 tile (tile_w x tile_h elements,
// row stride `stride` elements) from global into LDS at byte offset lds_off.
// Hardware pads the LDS destination: pad_interval=32 DWORDs (=128B = one
// 64-half row), pad_amount=4 DWORDs (=8 halves) -> effective LDS row stride
// of 72 halves, matching the bank-conflict-free [64][72] tiles below.
// Tracked on TENSORcnt; caller waits with s_wait_tensorcnt.
// ---------------------------------------------------------------------------
__device__ __forceinline__ void tdm_load_2d_f16(unsigned lds_off,
                                                const _Float16* gptr,
                                                unsigned tile_w, unsigned tile_h,
                                                unsigned long long stride) {
    const unsigned long long ga = (unsigned long long)(size_t)gptr;
    u32x4 g0;
    g0[0] = 1u;                                        // count=1, user mode
    g0[1] = lds_off;                                   // lds_addr
    g0[2] = (unsigned)(ga & 0xFFFFFFFFull);            // global_addr[31:0]
    g0[3] = (unsigned)((ga >> 32) & 0x1FFFFFFull)      // global_addr[56:32]
          | (2u << 30);                                // type=2 ("image")
    union { unsigned long long u64[4]; int i32[8]; } g1;
    const unsigned long long flags =
        (1ull << 16)      // data_size = 2 bytes
      | (1ull << 20)      // pad_enable
      | (4ull << 22)      // pad_interval: 32 DWORDs
      | (3ull << 25);     // pad_amount : 4 DWORDs
    g1.u64[0] = flags | ((unsigned long long)(tile_w & 0xFFFFu) << 48);  // tensor_dim0[15:0]
    g1.u64[1] = ((unsigned long long)tile_w >> 16)                        // tensor_dim0[31:16]
              | ((unsigned long long)tile_h << 16)                        // tensor_dim1
              | ((unsigned long long)tile_w << 48);                       // tile_dim0
    g1.u64[2] = (unsigned long long)tile_h                                // tile_dim1
              | ((stride & 0xFFFFFFFFull) << 32);                         // dim0_stride[31:0]
    g1.u64[3] = (stride >> 32);                                           // dim0_stride[47:32]
    i32x8 vg1 = { g1.i32[0], g1.i32[1], g1.i32[2], g1.i32[3],
                  g1.i32[4], g1.i32[5], g1.i32[6], g1.i32[7] };
    const i32x4 z4 = {0, 0, 0, 0};
#if defined(__clang_major__) && (__clang_major__ >= 23)
    const i32x8 z8 = {0, 0, 0, 0, 0, 0, 0, 0};
    __builtin_amdgcn_tensor_load_to_lds(g0, vg1, z4, z4, z8, 0);
#else
    __builtin_amdgcn_tensor_load_to_lds(g0, vg1, z4, z4, 0);
#endif
}

// ---------------------------------------------------------------------------
// C[M,N] = A[M,K] * B[N,K]^T.  A: float or _Float16; B: float; C: OutT.
// f16 WMMA compute, fp32 accumulate.  N,K compile-time for cheap addressing.
// ---------------------------------------------------------------------------
template <typename AT, typename OutT, int N, int K>
__global__ __launch_bounds__(256)
void gemm_nt(const AT* __restrict__ A, const float* __restrict__ B,
             OutT* __restrict__ C, int M) {
    __shared__ _Float16 As[128][40];   // [m][k], +8 pad (row=80B, 16B aligned)
    __shared__ _Float16 Bs[32][136];   // [k][n], +8 pad (row=272B, 16B aligned)

    const int t    = threadIdx.x;
    const int lane = t & 31;
    const int wave = t >> 5;
    const int wm   = wave & 1;
    const int wn   = wave >> 1;
    const int m0   = blockIdx.y * 128;
    const int n0   = blockIdx.x * 128;

    const v8f vzero = {0.f,0.f,0.f,0.f,0.f,0.f,0.f,0.f};
    v8f acc[4][2];
    #pragma unroll
    for (int i = 0; i < 4; ++i)
        #pragma unroll
        for (int j = 0; j < 2; ++j) acc[i][j] = vzero;

    const int trow = t >> 3;        // 0..31
    const int tcol = (t & 7) * 4;   // 0,4,...,28

    for (int kk = 0; kk < K; kk += 32) {
        __syncthreads();
        if (kk + 32 < K) {
            __builtin_prefetch(A + (size_t)(m0 + (t & 127)) * K + kk + 32, 0, 1);
            __builtin_prefetch(B + (size_t)(n0 + (t & 127)) * K + kk + 32, 0, 1);
        }
        if constexpr (sizeof(AT) == 4) {
            #pragma unroll
            for (int p = 0; p < 4; ++p) {
                const int r = trow + p * 32;
                const float4 av = *(const float4*)(A + (size_t)(m0 + r) * K + kk + tcol);
                As[r][tcol + 0] = (_Float16)av.x;
                As[r][tcol + 1] = (_Float16)av.y;
                As[r][tcol + 2] = (_Float16)av.z;
                As[r][tcol + 3] = (_Float16)av.w;
            }
        } else {
            const int trh = t >> 2;            // 0..63
            const int tch = (t & 3) * 8;       // 0,8,16,24
            #pragma unroll
            for (int p = 0; p < 2; ++p) {
                const int r = trh + p * 64;
                const v8h av = *(const v8h*)((const _Float16*)A + (size_t)(m0 + r) * K + kk + tch);
                *(v8h*)&As[r][tch] = av;       // single ds_store_b128
            }
        }
        #pragma unroll
        for (int p = 0; p < 4; ++p) {
            const int r = trow + p * 32;
            const float4 bv = *(const float4*)(B + (size_t)(n0 + r) * K + kk + tcol);
            Bs[tcol + 0][r] = (_Float16)bv.x;
            Bs[tcol + 1][r] = (_Float16)bv.y;
            Bs[tcol + 2][r] = (_Float16)bv.z;
            Bs[tcol + 3][r] = (_Float16)bv.w;
        }
        __syncthreads();

        v16h bfr[2];
        #pragma unroll
        for (int j = 0; j < 2; ++j) {
            #pragma unroll
            for (int s = 0; s < 16; ++s)
                bfr[j][s] = Bs[(lane >> 4) * 8 + (s & 7) + (s >> 3) * 16][wn * 32 + j * 16 + (lane & 15)];
        }
        const int kb = (lane >> 4) * 8;
        #pragma unroll
        for (int i = 0; i < 4; ++i) {
            const int mrow = wm * 64 + i * 16 + (lane & 15);
            const v16h afr = frag16(&As[mrow][kb], &As[mrow][kb + 16]);
            #pragma unroll
            for (int j = 0; j < 2; ++j)
                acc[i][j] = wmma_f16(afr, bfr[j], acc[i][j]);
        }
    }

    const int hi = lane >> 4;
    const int ln = lane & 15;
    #pragma unroll
    for (int i = 0; i < 4; ++i) {
        const int mb = m0 + wm * 64 + i * 16 + hi * 8;
        #pragma unroll
        for (int j = 0; j < 2; ++j) {
            OutT* cp = C + (size_t)mb * N + n0 + wn * 32 + j * 16 + ln;
            #pragma unroll
            for (int r = 0; r < 8; ++r)
                cp[(size_t)r * N] = (OutT)acc[i][j][r];
        }
    }
}

// ---------------------------------------------------------------------------
// In-place RoPE on q,k thirds of the f16 qkv buffer.
// ---------------------------------------------------------------------------
__global__ __launch_bounds__(256)
void rope_kernel(_Float16* __restrict__ qkv, const int* __restrict__ pos, int npairs) {
    const int idx = blockIdx.x * blockDim.x + threadIdx.x;
    if (idx >= npairs) return;
    const int i     = idx & 31;
    const int h     = (idx >> 5) & 15;
    const int which = (idx >> 9) & 1;
    const int tok   = idx >> 10;
    const float p   = (float)pos[tok];
    const float inv = __expf(-0.28782313662f * (float)i);   // theta^(-2i/64)
    float sn, cs;
    __sincosf(p * inv, &sn, &cs);
    const size_t base = (size_t)tok * 3072 + which * 1024 + h * 64 + 2 * i;
    const float x1 = (float)qkv[base];
    const float x2 = (float)qkv[base + 1];
    qkv[base]     = (_Float16)(x1 * cs - x2 * sn);
    qkv[base + 1] = (_Float16)(x1 * sn + x2 * cs);
}

// ---------------------------------------------------------------------------
// Flash attention, operand-swapped: S^T = K*Q^T so K/V tiles stay in natural
// [key][dk] layout and stream in via double-buffered TDM async DMA.  The S^T
// C-layout is register-compatible with the A-layout needed for P*V, so P is
// rebuilt entirely in-lane (no LDS restage).  One block per (128 q, head, b).
// ---------------------------------------------------------------------------
__global__ __launch_bounds__(256)
void flash_attn_kernel(const _Float16* __restrict__ qkv, _Float16* __restrict__ attn_out) {
    __shared__ _Float16 QsT[64][136];     // [dk][q row] (transposed, q pre-scaled)
    __shared__ _Float16 Ks[2][64][72];    // [key][dk], TDM-padded rows
    __shared__ _Float16 Vs[2][64][72];

    const int t    = threadIdx.x;
    const int lane = t & 31;
    const int wave = t >> 5;
    const int hi   = lane >> 4;
    const int ln   = lane & 15;
    const int qt   = blockIdx.x;
    const int h    = blockIdx.y;
    const int b    = blockIdx.z;
    const int q0   = qt * 128;
    const int nkt  = qt * 2 + 2;          // causal: tiles up to the diagonal

    const _Float16* kv_base = qkv + (size_t)b * S_LEN * 3072 + 1024 + h * 64;

    // kick off tile 0 DMA (overlaps with Q staging below)
    if (wave == 0) {
        const _Float16* kp = kv_base;     // k0 = 0
        tdm_load_2d_f16((unsigned)(size_t)&Ks[0][0][0], kp,        64, 64, 3072ull);
        tdm_load_2d_f16((unsigned)(size_t)&Vs[0][0][0], kp + 1024, 64, 64, 3072ull);
    }

    // ---- stage Q^T (f16), folding in the 1/sqrt(64) softmax scale ----
    #pragma unroll
    for (int p = 0; p < 4; ++p) {
        const int chunk = t + p * 256;            // 1024 chunks of 8 halves
        const int row   = chunk >> 3;             // q row 0..127
        const int c8    = (chunk & 7) * 8;        // dk 0..56
        const v8h qv = *(const v8h*)(qkv + (size_t)(b * S_LEN + q0 + row) * 3072 + h * 64 + c8);
        #pragma unroll
        for (int e = 0; e < 8; ++e)
            QsT[c8 + e][row] = (_Float16)((float)qv[e] * 0.125f);
    }
    __syncthreads();

    // ---- per-wave Q^T B-fragments (lane = q column, elements = dk rows) ----
    v16h qb[2];
    #pragma unroll
    for (int ks = 0; ks < 2; ++ks) {
        #pragma unroll
        for (int s = 0; s < 16; ++s)
            qb[ks][s] = QsT[ks * 32 + hi * 8 + (s & 7) + (s >> 3) * 16][wave * 16 + ln];
    }

    const v8f vzero = {0.f,0.f,0.f,0.f,0.f,0.f,0.f,0.f};
    v8f oacc[4];
    #pragma unroll
    for (int n = 0; n < 4; ++n) oacc[n] = vzero;
    const int q_lane = q0 + wave * 16 + ln;       // this lane's query row
    float mold = -1e30f, lsum = 0.f;

    for (int kt = 0; kt < nkt; ++kt) {
        const int buf = kt & 1;
        const int k0  = kt * 64;
        __syncthreads();   // everyone done reading buf^1 before DMA overwrites it
        if (wave == 0) {
            if (kt + 1 < nkt) {
                const _Float16* kp = kv_base + (size_t)(k0 + 64) * 3072;
                tdm_load_2d_f16((unsigned)(size_t)&Ks[buf ^ 1][0][0], kp,        64, 64, 3072ull);
                tdm_load_2d_f16((unsigned)(size_t)&Vs[buf ^ 1][0][0], kp + 1024, 64, 64, 3072ull);
                __builtin_amdgcn_s_wait_tensorcnt(2);   // tile kt done (in-order)
            } else {
                __builtin_amdgcn_s_wait_tensorcnt(0);
            }
        }
        __syncthreads();   // tile kt visible to all waves

        // ---- S^T = K * Q^T : 64 keys x 16 q per wave ----
        v8f st[4];
        #pragma unroll
        for (int j = 0; j < 4; ++j) {
            v8f a = vzero;
            #pragma unroll
            for (int ks = 0; ks < 2; ++ks) {
                const int kb = hi * 8;
                const _Float16* ap = &Ks[buf][j * 16 + ln][ks * 32 + kb];
                a = wmma_f16(frag16(ap, ap + 16), qb[ks], a);
            }
            st[j] = a;
        }

        // ---- online softmax; stats per q live in-lane (one shfl_xor 16) ----
        const bool need_mask = (k0 + 63) > (q0 + wave * 16);
        float tmax = -1e30f;
        #pragma unroll
        for (int j = 0; j < 4; ++j) {
            #pragma unroll
            for (int r = 0; r < 8; ++r) {
                float v = st[j][r];
                if (need_mask && (k0 + j * 16 + hi * 8 + r) > q_lane) v = -1e30f;
                st[j][r] = v;
                tmax = fmaxf(tmax, v);
            }
        }
        tmax = fmaxf(tmax, __shfl_xor(tmax, 16, 32));
        const float mnew = fmaxf(mold, tmax);
        const float corr = __expf(mold - mnew);
        float rsum = 0.f;
        #pragma unroll
        for (int j = 0; j < 4; ++j)
            #pragma unroll
            for (int r = 0; r < 8; ++r) {
                const float e = __expf(st[j][r] - mnew);
                st[j][r] = e;
                rsum += e;
            }
        rsum += __shfl_xor(rsum, 16, 32);
        lsum = lsum * corr + rsum;
        mold = mnew;

        // rescale O (its rows are q = r + 8*hi; fetch that q's corr by shuffle)
        #pragma unroll
        for (int r = 0; r < 8; ++r) {
            const float cr = __shfl(corr, r + 8 * hi, 32);
            #pragma unroll
            for (int n = 0; n < 4; ++n) oacc[n][r] *= cr;
        }

        // ---- P fragments rebuilt in-lane from S^T registers ----
        v16h pa[2];
        #pragma unroll
        for (int ks = 0; ks < 2; ++ks) {
            union { v16h v; _Float16 e[16]; } pu;
            #pragma unroll
            for (int r = 0; r < 8; ++r) {
                pu.e[r]     = (_Float16)st[2 * ks][r];
                pu.e[8 + r] = (_Float16)st[2 * ks + 1][r];
            }
            pa[ks] = pu.v;
        }

        // ---- O += P * V ----
        #pragma unroll
        for (int n = 0; n < 4; ++n)
            #pragma unroll
            for (int ks = 0; ks < 2; ++ks) {
                v16h vb;
                #pragma unroll
                for (int s = 0; s < 16; ++s)
                    vb[s] = Vs[buf][ks * 32 + hi * 8 + (s & 7) + (s >> 3) * 16][n * 16 + ln];
                oacc[n] = wmma_f16(pa[ks], vb, oacc[n]);
            }
    }

    // ---- epilogue: O /= l, store f16 [b][s][h*64+dk] ----
    #pragma unroll
    for (int r = 0; r < 8; ++r) {
        const float lr   = __shfl(lsum, r + 8 * hi, 32);
        const float rinv = 1.0f / lr;
        const int   qr   = q0 + wave * 16 + r + hi * 8;
        #pragma unroll
        for (int n = 0; n < 4; ++n)
            attn_out[(size_t)(b * S_LEN + qr) * 1024 + h * 64 + n * 16 + ln] =
                (_Float16)(oacc[n][r] * rinv);
    }
}

// ---------------------------------------------------------------------------
extern "C" void kernel_launch(void* const* d_in, const int* in_sizes, int n_in,
                              void* d_out, int out_size, void* d_ws, size_t ws_size,
                              hipStream_t stream) {
    (void)in_sizes; (void)n_in; (void)out_size; (void)ws_size;
    const float* x     = (const float*)d_in[0];
    const int*   tpos  = (const int*)d_in[1];
    const float* w_qkv = (const float*)d_in[2];
    const float* w_o   = (const float*)d_in[3];
    float* out = (float*)d_out;

    _Float16* qkv  = (_Float16*)d_ws;                         // [8192][3072] f16
    _Float16* attn = qkv + (size_t)BATCH * S_LEN * 3 * DM;    // [8192][1024] f16

    const int M = BATCH * S_LEN;   // 8192
    dim3 blk(256);

    // 1) qkv = x @ w_qkv^T   (f32 in, f16 out)
    gemm_nt<float, _Float16, 3 * DM, DM>
        <<<dim3((3 * DM) / 128, M / 128), blk, 0, stream>>>(x, w_qkv, qkv, M);

    // 2) RoPE on q,k (in place, f16)
    const int npairs = M * 2 * NH * (DKH / 2);
    rope_kernel<<<(npairs + 255) / 256, blk, 0, stream>>>(qkv, tpos, npairs);

    // 3) causal flash attention (TDM-streamed K/V)
    flash_attn_kernel<<<dim3(S_LEN / 128, NH, BATCH), blk, 0, stream>>>(qkv, attn);

    // 4) out = attn @ w_o^T   (f16 in, f32 out)
    gemm_nt<_Float16, float, DM, DM>
        <<<dim3(DM / 128, M / 128), blk, 0, stream>>>(attn, w_o, out, M);
}
